// MUSTANG_Classifier_90933047590935
// MI455X (gfx1250) — compile-verified, matrix operand order checked by reading, not run on hardware
//
#include <hip/hip_runtime.h>
#include <hip/hip_bf16.h>
#include <math.h>

typedef __attribute__((ext_vector_type(16))) _Float16 v16h;
typedef __attribute__((ext_vector_type(8)))  _Float16 v8h;
typedef __attribute__((ext_vector_type(8)))  float    v8f;
typedef __attribute__((ext_vector_type(4)))  unsigned u32x4;
typedef __attribute__((ext_vector_type(8)))  int      i32x8;
typedef __attribute__((ext_vector_type(4)))  int      i32x4;

#if defined(__gfx1250__) && __has_builtin(__builtin_amdgcn_tensor_load_to_lds) && __has_builtin(__builtin_amdgcn_s_wait_tensorcnt)
#define USE_TDM 1
#else
#define USE_TDM 0
#endif

static constexpr int N0   = 20000;
static constexpr int EDG  = 160000;
static constexpr int F_IN = 1024;
static constexpr int HID  = 256;
static constexpr int EMB  = 64;
static constexpr int WLEN = 16;
static constexpr int NCLS = 4;
static constexpr float NEG_SLOPE = 0.2f;

// ---------- helpers ----------
__device__ __forceinline__ unsigned enc_f32(float f) {
  unsigned b = __float_as_uint(f);
  return (b & 0x80000000u) ? ~b : (b ^ 0x80000000u);
}
__device__ __forceinline__ float dec_u32(unsigned u) {
  unsigned b = (u & 0x80000000u) ? (u ^ 0x80000000u) : ~u;
  return __uint_as_float(b);
}

// ---------- generic small kernels ----------
__global__ void fill_f32_kernel(float* p, float v, int n) {
  int i = blockIdx.x * blockDim.x + threadIdx.x;
  if (i < n) p[i] = v;
}
__global__ void fill_u32_kernel(unsigned* p, unsigned v, int n) {
  int i = blockIdx.x * blockDim.x + threadIdx.x;
  if (i < n) p[i] = v;
}
__global__ void cvt_f16_kernel(const float* __restrict__ src, _Float16* __restrict__ dst, int n) {
  int i = blockIdx.x * blockDim.x + threadIdx.x;
  if (i < n) dst[i] = (_Float16)src[i];
}

// n_emb[n,:] = node_table[node_attr[n],:]
__global__ void gather_nemb_kernel(const float* __restrict__ table, const int* __restrict__ attr,
                                   float* __restrict__ ne, int Nn) {
  int i = blockIdx.x * blockDim.x + threadIdx.x;
  if (i >= Nn * EMB) return;
  int n = i / EMB, j = i % EMB;
  ne[i] = table[attr[n] * EMB + j];
}

// T = edge_table (10 x 64) @ We (64 x 512)   -> 10 x 512
__global__ void compute_T_kernel(const float* __restrict__ tab, const float* __restrict__ We,
                                 float* __restrict__ T, int rows) {
  int i = blockIdx.x * blockDim.x + threadIdx.x;
  if (i >= rows * 512) return;
  int r = i / 512, c = i % 512;
  float s = 0.f;
  for (int k = 0; k < EMB; ++k) s += tab[r * EMB + k] * We[k * 512 + c];
  T[i] = s;
}

// h16 = f16(concat(x, rwpe, n_emb))  (row major, dm columns)
__global__ void build_h_kernel(_Float16* __restrict__ h, const float* __restrict__ x, int Fx,
                               const float* __restrict__ rw, const float* __restrict__ ne,
                               int Nn, int dm) {
  int i = blockIdx.x * blockDim.x + threadIdx.x;
  if (i >= Nn * dm) return;
  int n = i / dm, c = i % dm;
  float v;
  if (c < Fx)               v = x[(size_t)n * Fx + c];
  else if (c < Fx + WLEN)   v = rw[(size_t)n * WLEN + (c - Fx)];
  else                      v = ne[(size_t)n * EMB + (c - Fx - WLEN)];
  h[i] = (_Float16)v;
}

#if USE_TDM
// TDM 2D tile load: 128 rows x 32 f16 cols from row-major A(ldk) into LDS with
// 40-half (80B) pitch via pad (4 DWORDs every 16 DWORDs). OOB -> zero fill.
__device__ __forceinline__ void tdm_load_tileA(unsigned lds_off, const _Float16* g,
                                               int rem_k, int rem_m, int ldk) {
  unsigned long long ga = (unsigned long long)(uintptr_t)g;
  u32x4 g0;
  g0.x = 1u;                                   // count=1, user descriptor
  g0.y = lds_off;                              // lds byte address
  g0.z = (unsigned)(ga & 0xffffffffull);       // global_addr[31:0] -> bits 95:64
  g0.w = (unsigned)((ga >> 32) & 0x01ffffffull) | (2u << 30);  // addr[56:32], type=2
  unsigned td0 = (unsigned)rem_k;
  unsigned td1 = (unsigned)rem_m;
  unsigned long long st0 = (unsigned long long)(unsigned)ldk;  // dim0 stride (elems)
  i32x8 g1;
  g1[0] = (int)((1u << 16)                     // data_size = 2 bytes
                | (1u << 20)                   // pad_enable
                | (3u << 22)                   // pad_interval: 16 DWORDs
                | (3u << 25));                 // pad_amount: 4 DWORDs
  g1[1] = (int)((td0 & 0xffffu) << 16);        // tensor_dim0[15:0] @ bits 63:48
  g1[2] = (int)((td0 >> 16) | ((td1 & 0xffffu) << 16));  // td0 hi | td1 lo
  g1[3] = (int)((td1 >> 16) | (32u << 16));    // td1 hi | tile_dim0=32
  g1[4] = (int)128u;                           // tile_dim1=128 (tile_dim2=0)
  g1[5] = (int)(unsigned)(st0 & 0xffffffffull);          // stride0[31:0]
  g1[6] = (int)(unsigned)((st0 >> 32) & 0xffffull);      // stride0[47:32]
  g1[7] = 0;
  i32x4 z4 = {0, 0, 0, 0};
  i32x8 z8 = {0, 0, 0, 0, 0, 0, 0, 0};
  __builtin_amdgcn_tensor_load_to_lds(g0, g1, z4, z4, z8, 0);
}
#endif

// build v16h WMMA fragment from a 40-half-pitch LDS row: two 16B vector loads
__device__ __forceinline__ v16h frag16(const _Float16* rowp, int base) {
  v8h lo = *(const v8h*)(rowp + base);
  v8h hv = *(const v8h*)(rowp + base + 16);
  return __builtin_shufflevector(lo, hv, 0, 1, 2, 3, 4, 5, 6, 7,
                                 8, 9, 10, 11, 12, 13, 14, 15);
}

// ---------- WMMA GEMM: C(MxN f32) = A(MxK f16) @ B(KxN f16) ----------
// Block: 256 threads = 8 waves, macro-tile 128(M) x 64(N); each wave owns a
// 2x2 grid of 16x16 D tiles (4 WMMAs per K=32 step).
__global__ __launch_bounds__(256)
void gemm_f16_wmma_kernel(const _Float16* __restrict__ A, const _Float16* __restrict__ B,
                          float* __restrict__ C, int M, int K, int N) {
  __shared__ __align__(16) _Float16 As[128][40];
  __shared__ __align__(16) _Float16 Bst[64][40];   // transposed: Bst[n][k]
  const int tid  = threadIdx.x;
  const int lane = tid & 31;
  const int w    = tid >> 5;
  const int wm   = w & 3;        // 4 wave-tiles along M (32 rows each)
  const int wn   = w >> 2;       // 2 wave-tiles along N (32 cols each)
  const int lr   = lane & 15;
  const int hi   = lane >> 4;
  const int m0   = blockIdx.y * 128;
  const int n0   = blockIdx.x * 64;
  v8f acc00 = {}, acc01 = {}, acc10 = {}, acc11 = {};
  for (int k0 = 0; k0 < K; k0 += 32) {
#if USE_TDM
    if (w == 0) {
      tdm_load_tileA((unsigned)(uintptr_t)&As[0][0], A + (size_t)m0 * K + k0,
                     K - k0, M - m0, K);
      __builtin_amdgcn_s_wait_tensorcnt(0);
    }
#else
    for (int t = tid; t < 128 * 32; t += 256) {
      int r = t >> 5, c = t & 31;
      int gr = m0 + r, gc = k0 + c;
      As[r][c] = (gr < M && gc < K) ? A[(size_t)gr * K + gc] : (_Float16)0.f;
    }
#endif
    // stage B transposed (coalesced along N)
    for (int t = tid; t < 32 * 64; t += 256) {
      int c = t & 63, r = t >> 6;
      int gr = k0 + r, gc = n0 + c;
      Bst[c][r] = (gr < K && gc < N) ? B[(size_t)gr * N + gc] : (_Float16)0.f;
    }
    if (k0 + 32 < K && tid < 64 && (m0 + tid) < M)
      __builtin_prefetch(&A[(size_t)(m0 + tid) * K + k0 + 32], 0, 1);
    __syncthreads();
    // fragments per CDNA5 16-bit lane layout (ISA 7.12.2): contiguous 8-half runs
    const int base = hi * 8;
    const int ar0 = wm * 32 + lr, ar1 = ar0 + 16;
    const int bc0 = wn * 32 + lr, bc1 = bc0 + 16;
    v16h a0 = frag16(&As[ar0][0], base);
    v16h a1 = frag16(&As[ar1][0], base);
    v16h b0 = frag16(&Bst[bc0][0], base);
    v16h b1 = frag16(&Bst[bc1][0], base);
    acc00 = __builtin_amdgcn_wmma_f32_16x16x32_f16(false, a0, false, b0, (short)0, acc00, false, false);
    acc01 = __builtin_amdgcn_wmma_f32_16x16x32_f16(false, a0, false, b1, (short)0, acc01, false, false);
    acc10 = __builtin_amdgcn_wmma_f32_16x16x32_f16(false, a1, false, b0, (short)0, acc10, false, false);
    acc11 = __builtin_amdgcn_wmma_f32_16x16x32_f16(false, a1, false, b1, (short)0, acc11, false, false);
    __syncthreads();
  }
  const int colb = n0 + wn * 32 + lr;
#pragma unroll
  for (int r = 0; r < 8; ++r) {
    int row0 = m0 + wm * 32 + hi * 8 + r;
    int row1 = row0 + 16;
    if (row0 < M && colb < N)      C[(size_t)row0 * N + colb]      = acc00[r];
    if (row0 < M && colb + 16 < N) C[(size_t)row0 * N + colb + 16] = acc01[r];
    if (row1 < M && colb < N)      C[(size_t)row1 * N + colb]      = acc10[r];
    if (row1 < M && colb + 16 < N) C[(size_t)row1 * N + colb + 16] = acc11[r];
  }
}

// ---------- edge phase ----------
__global__ __launch_bounds__(256)
void edge_logit_kernel(const float* __restrict__ XL, const float* __restrict__ XR,
                       const float* __restrict__ T, const float* __restrict__ att,
                       const int* __restrict__ ea, const int* __restrict__ src,
                       const int* __restrict__ dst, const float* __restrict__ emask,
                       float* __restrict__ logit, unsigned* __restrict__ mx) {
  int i = blockIdx.x * blockDim.x + threadIdx.x;
  if (i >= 2 * EDG) return;
  int h = (i < EDG) ? 0 : 1;
  int e = i - h * EDG;
  int s = src[e], d = dst[e];
  const float* xl = XL + (size_t)s * 512 + h * 256;
  const float* xr = XR + (size_t)d * 512 + h * 256;
  const float* te = T + (size_t)ea[e] * 512 + h * 256;
  const float* at = att + h * 256;
  float acc = 0.f;
  for (int j = 0; j < 256; ++j) {
    float m = xl[j] + xr[j] + te[j];
    m = (m > 0.f) ? m : NEG_SLOPE * m;
    acc += m * at[j];
  }
  float lg = (emask[e] > 0.f) ? acc : -1e9f;
  logit[(size_t)e * 2 + h] = lg;
  atomicMax(&mx[d * 2 + h], enc_f32(lg));
}

__global__ __launch_bounds__(256)
void edge_p_kernel(float* __restrict__ logit, const unsigned* __restrict__ mx,
                   float* __restrict__ den, const int* __restrict__ dst,
                   const float* __restrict__ emask) {
  int i = blockIdx.x * blockDim.x + threadIdx.x;
  if (i >= 2 * EDG) return;
  int h = (i < EDG) ? 0 : 1;
  int e = i - h * EDG;
  int d = dst[e];
  float lg = logit[(size_t)e * 2 + h];
  float m  = dec_u32(mx[d * 2 + h]);
  float p  = expf(lg - m) * emask[e];
  logit[(size_t)e * 2 + h] = p;
  if (p != 0.f) atomicAdd(&den[d * 2 + h], p);
}

__global__ __launch_bounds__(256)
void edge_agg_kernel(const float* __restrict__ p, const float* __restrict__ den,
                     const float* __restrict__ XL, float* __restrict__ AG,
                     const int* __restrict__ src, const int* __restrict__ dst) {
  int i = blockIdx.x * blockDim.x + threadIdx.x;
  if (i >= 2 * EDG) return;
  int h = (i < EDG) ? 0 : 1;
  int e = i - h * EDG;
  float pv = p[(size_t)e * 2 + h];
  if (pv == 0.f) return;
  int s = src[e], d = dst[e];
  float alpha = pv / (den[d * 2 + h] + 1e-16f);
  const float* xl = XL + (size_t)s * 512 + h * 256;
  float* ag = AG + (size_t)d * 512 + h * 256;
  for (int j = 0; j < 256; ++j) atomicAdd(&ag[j], alpha * xl[j]);
}

__global__ __launch_bounds__(256)
void node_update_kernel(const float* __restrict__ AG, const float* __restrict__ bias,
                        float* __restrict__ xnew, const float* __restrict__ pw_rel,
                        const float* __restrict__ pw_root, float* __restrict__ rel,
                        float* __restrict__ root) {
  int n = blockIdx.x, j = threadIdx.x;
  float v = 0.5f * (AG[(size_t)n * 512 + j] + AG[(size_t)n * 512 + 256 + j]) + bias[j];
  v = (v > 0.f) ? v : 0.f;
  xnew[(size_t)n * 256 + j] = v;
  __shared__ float red[256];
  red[j] = v * pw_rel[j];
  __syncthreads();
  for (int s = 128; s > 0; s >>= 1) { if (j < s) red[j] += red[j + s]; __syncthreads(); }
  if (j == 0) rel[n] = red[0];
  __syncthreads();
  red[j] = v * pw_root[j];
  __syncthreads();
  for (int s = 128; s > 0; s >>= 1) { if (j < s) red[j] += red[j + s]; __syncthreads(); }
  if (j == 0) root[n] = red[0];
}

__global__ void edge_neigh_kernel(const int* __restrict__ src, const int* __restrict__ dst,
                                  const float* __restrict__ emask, const float* __restrict__ rel,
                                  float* __restrict__ neigh) {
  int e = blockIdx.x * blockDim.x + threadIdx.x;
  if (e >= EDG) return;
  if (emask[e] > 0.f) atomicAdd(&neigh[dst[e]], rel[src[e]]);
}

__global__ void score_key_kernel(const float* __restrict__ root, const float* __restrict__ neigh,
                                 const float* __restrict__ pb, float* __restrict__ score,
                                 unsigned* __restrict__ keys, int Nn) {
  int n = blockIdx.x * blockDim.x + threadIdx.x;
  if (n >= Nn) return;
  float s = tanhf(root[n] + neigh[n] + pb[0]);
  score[n] = s;
  keys[n] = enc_f32(s);
}

// ---------- device-side radix top-k select ----------
__global__ void topk_init_kernel(unsigned* st, unsigned k) {
  st[0] = 0u; st[1] = k; st[2] = 0u; st[3] = 0u;
}
__global__ void topk_hist_kernel(const unsigned* __restrict__ keys, int Nn, int b,
                                 const unsigned* __restrict__ st, unsigned* __restrict__ hist) {
  int n = blockIdx.x * blockDim.x + threadIdx.x;
  if (n >= Nn) return;
  unsigned key = keys[n];
  unsigned hb = (b == 3) ? 0u : (key >> ((b + 1) * 8));
  if (hb == st[0]) atomicAdd(&hist[(key >> (b * 8)) & 255u], 1u);
}
__global__ void topk_scan_kernel(const unsigned* __restrict__ hist, unsigned* st) {
  unsigned prefix = st[0];
  unsigned kneed = st[1];
  unsigned cum = 0, chosen = 0;
  for (int d = 255; d >= 0; --d) {
    unsigned c = hist[d];
    if (cum + c >= kneed) { chosen = (unsigned)d; break; }
    cum += c;
  }
  st[0] = (prefix << 8) | chosen;
  st[1] = kneed - cum;
  st[2] = 0u; st[3] = 0u;
}
__global__ void topk_select_kernel(const unsigned* __restrict__ keys, unsigned* st,
                                   int* __restrict__ new_idx, int Nn) {
  int n = blockIdx.x * blockDim.x + threadIdx.x;
  if (n >= Nn) return;
  unsigned key = keys[n], thr = st[0];
  bool sel = false;
  if (key > thr) sel = true;
  else if (key == thr) { unsigned t = atomicAdd(&st[3], 1u); sel = (t < st[1]); }
  new_idx[n] = sel ? (int)atomicAdd(&st[2], 1u) : -1;
}

__global__ __launch_bounds__(256)
void gather_kernel(const int* __restrict__ new_idx, const float* __restrict__ score,
                   const float* __restrict__ xnew, const float* __restrict__ rw_cur,
                   const float* __restrict__ ne_cur, float* __restrict__ x_nxt,
                   float* __restrict__ rw_nxt, float* __restrict__ ne_nxt) {
  int n = blockIdx.x;
  int pos = new_idx[n];
  if (pos < 0) return;
  int j = threadIdx.x;
  float sc = score[n];
  x_nxt[(size_t)pos * 256 + j] = xnew[(size_t)n * 256 + j] * sc;
  if (j < WLEN) rw_nxt[(size_t)pos * WLEN + j] = rw_cur[(size_t)n * WLEN + j];
  if (j < EMB)  ne_nxt[(size_t)pos * EMB + j]  = ne_cur[(size_t)n * EMB + j];
}

__global__ void remap_edges_kernel(const int* __restrict__ new_idx, int* __restrict__ src,
                                   int* __restrict__ dst, float* __restrict__ emask) {
  int e = blockIdx.x * blockDim.x + threadIdx.x;
  if (e >= EDG) return;
  int s2 = new_idx[src[e]];
  int d2 = new_idx[dst[e]];
  bool keep = (s2 >= 0) && (d2 >= 0);
  emask[e] = keep ? emask[e] : 0.f;
  src[e] = keep ? s2 : 0;
  dst[e] = keep ? d2 : 0;
}

__global__ __launch_bounds__(256)
void pool_kernel(const float* __restrict__ x, int k, float* __restrict__ pooled) {
  int c = blockIdx.x, t = threadIdx.x;
  float sm = 0.f, mx = -1e38f;
  for (int r = t; r < k; r += 256) {
    float v = x[(size_t)r * 256 + c];
    sm += v;
    mx = fmaxf(mx, v);
  }
  __shared__ float ss[256], sx[256];
  ss[t] = sm; sx[t] = mx;
  __syncthreads();
  for (int s = 128; s > 0; s >>= 1) {
    if (t < s) { ss[t] += ss[t + s]; sx[t] = fmaxf(sx[t], sx[t + s]); }
    __syncthreads();
  }
  if (t == 0) {
    pooled[c] += ss[0] / (float)k;
    pooled[256 + c] += sx[0];
  }
}

// ---------- MLP head + softmax ----------
__global__ __launch_bounds__(256)
void head_kernel(const float* __restrict__ pooled, const float* __restrict__ w1,
                 const float* __restrict__ b1, const float* __restrict__ w2,
                 const float* __restrict__ b2, const float* __restrict__ w3,
                 const float* __restrict__ b3, const int* __restrict__ label,
                 float* __restrict__ out) {
  __shared__ float h1[256];
  __shared__ float h2[128];
  __shared__ float lg[NCLS];
  int t = threadIdx.x;
  {
    float s = b1[t];
    for (int i = 0; i < 512; ++i) s += pooled[i] * w1[i * 256 + t];
    h1[t] = (s > 0.f) ? s : 0.f;
  }
  __syncthreads();
  if (t < 128) {
    float s = b2[t];
    for (int i = 0; i < 256; ++i) s += h1[i] * w2[i * 128 + t];
    h2[t] = (s > 0.f) ? s : 0.f;
  }
  __syncthreads();
  if (t < NCLS) {
    float s = b3[t];
    for (int i = 0; i < 128; ++i) s += h2[i] * w3[i * NCLS + t];
    lg[t] = s;
  }
  __syncthreads();
  if (t == 0) {
    float mx = lg[0];
    for (int i = 1; i < NCLS; ++i) mx = fmaxf(mx, lg[i]);
    float den = 0.f, e[NCLS];
    for (int i = 0; i < NCLS; ++i) { e[i] = expf(lg[i] - mx); den += e[i]; }
    for (int i = 0; i < NCLS; ++i) { out[i] = lg[i]; out[NCLS + i] = e[i] / den; }
    out[2 * NCLS] = (float)label[0];
  }
}

// ---------- host orchestration ----------
static inline char* ws_take(char*& p, size_t bytes) {
  char* r = p;
  p += (bytes + 255) & ~((size_t)255);
  return r;
}

extern "C" void kernel_launch(void* const* d_in, const int* in_sizes, int n_in,
                              void* d_out, int out_size, void* d_ws, size_t ws_size,
                              hipStream_t stream) {
  (void)in_sizes; (void)n_in; (void)out_size; (void)ws_size;
  const float* x_in   = (const float*)d_in[0];
  const float* rw_in  = (const float*)d_in[1];
  const float* b1     = (const float*)d_in[2];
  const float* b2     = (const float*)d_in[3];
  const float* b3     = (const float*)d_in[4];
  const float* etab   = (const float*)d_in[5];
  // layers at 6 + l*8 : We, Wl, Wr, att, bias, pb, pw_rel, pw_root
  const float* ntab   = (const float*)d_in[38];
  const float* w1     = (const float*)d_in[39];
  const float* w2     = (const float*)d_in[40];
  const float* w3     = (const float*)d_in[41];
  const int* edge_attr = (const int*)d_in[42];
  const int* node_attr = (const int*)d_in[43];
  const int* edge_src  = (const int*)d_in[44];
  const int* edge_dst  = (const int*)d_in[45];
  const int* label     = (const int*)d_in[46];

  char* p = (char*)d_ws;
  _Float16* h16  = (_Float16*)ws_take(p, (size_t)N0 * 1104 * 2);
  _Float16* Wl16 = (_Float16*)ws_take(p, (size_t)1104 * 512 * 2);
  _Float16* Wr16 = (_Float16*)ws_take(p, (size_t)1104 * 512 * 2);
  float* T     = (float*)ws_take(p, (size_t)10 * 512 * 4);
  float* XL    = (float*)ws_take(p, (size_t)N0 * 512 * 4);
  float* XR    = (float*)ws_take(p, (size_t)N0 * 512 * 4);   // reused as agg
  float* logit = (float*)ws_take(p, (size_t)EDG * 2 * 4);    // reused as p
  unsigned* mx = (unsigned*)ws_take(p, (size_t)N0 * 2 * 4);
  float* den   = (float*)ws_take(p, (size_t)N0 * 2 * 4);
  float* xA    = (float*)ws_take(p, (size_t)N0 * 256 * 4);
  float* xB    = (float*)ws_take(p, (size_t)N0 * 256 * 4);
  float* rwA   = (float*)ws_take(p, (size_t)N0 * WLEN * 4);
  float* rwB   = (float*)ws_take(p, (size_t)N0 * WLEN * 4);
  float* neA   = (float*)ws_take(p, (size_t)N0 * EMB * 4);
  float* neB   = (float*)ws_take(p, (size_t)N0 * EMB * 4);
  int* srcW    = (int*)ws_take(p, (size_t)EDG * 4);
  int* dstW    = (int*)ws_take(p, (size_t)EDG * 4);
  float* emaskW = (float*)ws_take(p, (size_t)EDG * 4);
  float* rel   = (float*)ws_take(p, (size_t)N0 * 4);
  float* root  = (float*)ws_take(p, (size_t)N0 * 4);
  float* neigh = (float*)ws_take(p, (size_t)N0 * 4);
  float* score = (float*)ws_take(p, (size_t)N0 * 4);
  unsigned* keys = (unsigned*)ws_take(p, (size_t)N0 * 4);
  int* new_idx = (int*)ws_take(p, (size_t)N0 * 4);
  float* pooled = (float*)ws_take(p, (size_t)512 * 4);
  unsigned* hist = (unsigned*)ws_take(p, (size_t)256 * 4);
  unsigned* st   = (unsigned*)ws_take(p, (size_t)16 * 4);

  const int dms[4] = {1104, 336, 336, 336};
  const int Ns[4]  = {20000, 14000, 9800, 6860};
  const int Ksel[4] = {14000, 9800, 6860, 4802};

  (void)hipMemcpyAsync(srcW, edge_src, (size_t)EDG * 4, hipMemcpyDeviceToDevice, stream);
  (void)hipMemcpyAsync(dstW, edge_dst, (size_t)EDG * 4, hipMemcpyDeviceToDevice, stream);
  fill_f32_kernel<<<(EDG + 255) / 256, 256, 0, stream>>>(emaskW, 1.0f, EDG);
  (void)hipMemsetAsync(pooled, 0, 512 * 4, stream);
  gather_nemb_kernel<<<(N0 * EMB + 255) / 256, 256, 0, stream>>>(ntab, node_attr, neA, N0);

  const float* x_cur = x_in;
  const float* rw_cur = rw_in;
  const float* ne_cur = neA;

  for (int l = 0; l < 4; ++l) {
    const int li = 6 + l * 8;
    const float* We     = (const float*)d_in[li + 0];
    const float* Wl     = (const float*)d_in[li + 1];
    const float* Wr     = (const float*)d_in[li + 2];
    const float* att    = (const float*)d_in[li + 3];
    const float* bias   = (const float*)d_in[li + 4];
    const float* pb     = (const float*)d_in[li + 5];
    const float* pw_rel = (const float*)d_in[li + 6];
    const float* pw_rt  = (const float*)d_in[li + 7];
    const int Nn = Ns[l], dm = dms[l], kk = Ksel[l];
    const int Fx = (l == 0) ? F_IN : HID;
    float* x_nxt  = (l & 1) ? xB : xA;
    float* rw_nxt = (l & 1) ? rwB : rwA;
    float* ne_nxt = (l & 1) ? neA : neB;   // ne_cur starts at neA

    build_h_kernel<<<(Nn * dm + 255) / 256, 256, 0, stream>>>(h16, x_cur, Fx, rw_cur, ne_cur, Nn, dm);
    cvt_f16_kernel<<<(dm * 512 + 255) / 256, 256, 0, stream>>>(Wl, Wl16, dm * 512);
    cvt_f16_kernel<<<(dm * 512 + 255) / 256, 256, 0, stream>>>(Wr, Wr16, dm * 512);
    compute_T_kernel<<<(10 * 512 + 255) / 256, 256, 0, stream>>>(etab, We, T, 10);

    dim3 gg(512 / 64, (Nn + 127) / 128);
    gemm_f16_wmma_kernel<<<gg, 256, 0, stream>>>(h16, Wl16, XL, Nn, dm, 512);
    gemm_f16_wmma_kernel<<<gg, 256, 0, stream>>>(h16, Wr16, XR, Nn, dm, 512);

    fill_u32_kernel<<<(Nn * 2 + 255) / 256, 256, 0, stream>>>(mx, 0x007FFFFFu, Nn * 2);
    (void)hipMemsetAsync(den, 0, (size_t)Nn * 2 * 4, stream);
    edge_logit_kernel<<<(2 * EDG + 255) / 256, 256, 0, stream>>>(XL, XR, T, att, edge_attr, srcW, dstW, emaskW, logit, mx);
    edge_p_kernel<<<(2 * EDG + 255) / 256, 256, 0, stream>>>(logit, mx, den, dstW, emaskW);
    (void)hipMemsetAsync(XR, 0, (size_t)Nn * 512 * 4, stream);
    edge_agg_kernel<<<(2 * EDG + 255) / 256, 256, 0, stream>>>(logit, den, XL, XR, srcW, dstW);

    node_update_kernel<<<Nn, 256, 0, stream>>>(XR, bias, XL, pw_rel, pw_rt, rel, root);

    (void)hipMemsetAsync(neigh, 0, (size_t)Nn * 4, stream);
    edge_neigh_kernel<<<(EDG + 255) / 256, 256, 0, stream>>>(srcW, dstW, emaskW, rel, neigh);
    score_key_kernel<<<(Nn + 255) / 256, 256, 0, stream>>>(root, neigh, pb, score, keys, Nn);

    topk_init_kernel<<<1, 1, 0, stream>>>(st, (unsigned)kk);
    for (int b = 3; b >= 0; --b) {
      (void)hipMemsetAsync(hist, 0, 256 * 4, stream);
      topk_hist_kernel<<<(Nn + 255) / 256, 256, 0, stream>>>(keys, Nn, b, st, hist);
      topk_scan_kernel<<<1, 1, 0, stream>>>(hist, st);
    }
    topk_select_kernel<<<(Nn + 255) / 256, 256, 0, stream>>>(keys, st, new_idx, Nn);

    gather_kernel<<<Nn, 256, 0, stream>>>(new_idx, score, XL, rw_cur, ne_cur, x_nxt, rw_nxt, ne_nxt);
    remap_edges_kernel<<<(EDG + 255) / 256, 256, 0, stream>>>(new_idx, srcW, dstW, emaskW);
    pool_kernel<<<256, 256, 0, stream>>>(x_nxt, kk, pooled);

    x_cur = x_nxt; rw_cur = rw_nxt; ne_cur = ne_nxt;
  }

  head_kernel<<<1, 256, 0, stream>>>(pooled, w1, b1, w2, b2, w3, b3, label, (float*)d_out);
}
